// LocalConstructorMulti_72705206387198
// MI455X (gfx1250) — matrix operand order repaired and checked by reading, not compiled, not run on hardware
//
#include <hip/hip_runtime.h>
#include <hip/hip_bf16.h>

// ---------------- types ----------------
typedef __bf16          bf16_t;
typedef bf16_t          v16bf  __attribute__((ext_vector_type(16)));
typedef bf16_t          v8bf   __attribute__((ext_vector_type(8)));
typedef float           v8f    __attribute__((ext_vector_type(8)));
typedef float           f32x4  __attribute__((ext_vector_type(4)));

// ---------------- problem constants ----------------
constexpr int kHID   = 4096;
constexpr int kSLOTS = 8;
constexpr int kHEADS = 8;
constexpr int kBD    = 512;
constexpr int kHD    = 64;
constexpr int kB     = 4;
constexpr int kS     = 4096;
constexpr int kM     = kB * kS;      // 16384 rows of hidden_states

// GEMM tiling for the fused K/V projection
constexpr int MBLK  = 64;    // rows of H per workgroup
constexpr int NBLK  = 128;   // output cols per workgroup (of 512)
constexpr int KSTEP = 32;    // bf16 WMMA K
constexpr int NK    = kHID / KSTEP;   // 128 k-steps

// ============================================================
// Kernel 1: Q = memory_slots @ Wq^T   (8 x 512, tiny)
// ============================================================
__global__ __launch_bounds__(256) void qproj_kernel(
    const float* __restrict__ mem, const float* __restrict__ Wq,
    float* __restrict__ Qbuf) {
  int idx = blockIdx.x * 256 + threadIdx.x;   // 0..4095
  int n = idx >> 9, o = idx & 511;
  const float* a = mem + (size_t)n * kHID;
  const float* w = Wq  + (size_t)o * kHID;
  float acc = 0.f;
  for (int i = 0; i < kHID; i += 4) {
    f32x4 x = *(const f32x4*)(a + i);
    f32x4 y = *(const f32x4*)(w + i);
    acc += x[0]*y[0] + x[1]*y[1] + x[2]*y[2] + x[3]*y[3];
  }
  Qbuf[idx] = acc;   // idx == n*512 + o
}

// ============================================================
// Kernel 2: fused K/V projection with bf16 WMMA, fp32 accumulate
//   K = H @ Wk^T, V = H @ Wv^T  (16384x4096 @ 4096x512, NT gemm)
//   grid = (kM/MBLK, kBD/NBLK), block = 256 (8 waves)
//   wave w: M-subtile = w&3, N-subtiles = (w>>2)*4 .. +3, both K and V
//   Double-buffered LDS (ping-pong), 1 barrier per k-step; sched
//   barriers force: [stage next slab] [16x ds_load frags] [8x WMMA].
// ============================================================
__global__ __launch_bounds__(256) void kvproj_wmma_kernel(
    const float* __restrict__ H,  const float* __restrict__ Wk,
    const float* __restrict__ Wv, float* __restrict__ Kout,
    float* __restrict__ Vout) {
  __shared__ __align__(64) bf16_t lh [2][MBLK * KSTEP];   // 2 x 4 KB
  __shared__ __align__(64) bf16_t lwk[2][NBLK * KSTEP];   // 2 x 8 KB
  __shared__ __align__(64) bf16_t lwv[2][NBLK * KSTEP];   // 2 x 8 KB

  const int tid   = threadIdx.x;
  const int wave  = tid >> 5;
  const int lane  = tid & 31;
  const int lrow  = lane & 15;    // N (for B/C/D) or M (for A)
  const int lhalf = lane >> 4;    // lane-half selects K/M sub-range
  const int msub  = wave & 3;
  const int nbase = (wave >> 2) * 4;

  const int blockM = blockIdx.x * MBLK;
  const int blockN = blockIdx.y * NBLK;

  v8f accK[4] = {};
  v8f accV[4] = {};

  const int hr = tid >> 2, hseg = tid & 3;  // H loader: 8 floats/thread
  const int wr = tid >> 1, wseg = tid & 1;  // W loader: 16 floats/thread

  // global -> fp32 regs -> v_cvt_pk_bf16_f32 -> LDS (one slab of K=32)
  auto load_tile = [&](int k0, int p) {
    {
      const float* src = H + (size_t)(blockM + hr) * kHID + k0 + hseg * 8;
      f32x4 x0 = *(const f32x4*)src;
      f32x4 x1 = *(const f32x4*)(src + 4);
      bf16_t* dst = &lh[p][hr * KSTEP + hseg * 8];
      #pragma unroll
      for (int e = 0; e < 4; ++e) { dst[e] = (bf16_t)x0[e]; dst[4 + e] = (bf16_t)x1[e]; }
      if (k0 + KSTEP < kHID)
        __builtin_prefetch(src + KSTEP, 0, 1);   // global_prefetch next slab
    }
    {
      const float* sk = Wk + (size_t)(blockN + wr) * kHID + k0 + wseg * 16;
      const float* sv = Wv + (size_t)(blockN + wr) * kHID + k0 + wseg * 16;
      bf16_t* dk = &lwk[p][wr * KSTEP + wseg * 16];
      bf16_t* dv = &lwv[p][wr * KSTEP + wseg * 16];
      #pragma unroll
      for (int q = 0; q < 4; ++q) {
        f32x4 a = *(const f32x4*)(sk + q * 4);
        f32x4 b = *(const f32x4*)(sv + q * 4);
        #pragma unroll
        for (int e = 0; e < 4; ++e) { dk[q*4+e] = (bf16_t)a[e]; dv[q*4+e] = (bf16_t)b[e]; }
      }
    }
  };

  // prologue: fill buffer 0 with slab 0
  load_tile(0, 0);
  __syncthreads();

  int p = 0;
  for (int kk = 0; kk < NK; ++kk) {
    // stage next slab into the other buffer (overlaps with WMMAs below)
    if (kk + 1 < NK) load_tile((kk + 1) * KSTEP, p ^ 1);

    // ---- A fragment (16x32 bf16, contiguous-k order, not ISA table order) ----
    // lane<16: K = 0..15; lane>=16: K = 16..31 (contiguous, matches B)
    v16bf av = *(const v16bf*)&lh[p][(msub * 16 + lrow) * KSTEP + 16 * lhalf];

    // ---- load ALL 8 B fragments first (distinct registers) ----
    v16bf bk[4], bv[4];
    #pragma unroll
    for (int t = 0; t < 4; ++t) {
      const int nrow = (nbase + t) * 16 + lrow;
      bk[t] = *(const v16bf*)&lwk[p][nrow * KSTEP + 16 * lhalf];
      bv[t] = *(const v16bf*)&lwv[p][nrow * KSTEP + 16 * lhalf];
    }

    // pin: all ds_loads above, all WMMAs below -> single s_wait_dscnt,
    // then 8 back-to-back v_wmma (XDL-saturating issue pattern)
    __builtin_amdgcn_sched_barrier(0);

    #pragma unroll
    for (int t = 0; t < 4; ++t)
      accK[t] = __builtin_amdgcn_wmma_f32_16x16x32_bf16(
          false, av, false, bk[t], (short)0, accK[t], false, false);
    #pragma unroll
    for (int t = 0; t < 4; ++t)
      accV[t] = __builtin_amdgcn_wmma_f32_16x16x32_bf16(
          false, av, false, bv[t], (short)0, accV[t], false, false);

    __builtin_amdgcn_sched_barrier(0);

    __syncthreads();
    p ^= 1;
  }

  // ---- store C/D: VGPR r -> M = r + 8*lhalf, N = lrow ----
  #pragma unroll
  for (int t = 0; t < 4; ++t) {
    const int ncol = blockN + (nbase + t) * 16 + lrow;
    #pragma unroll
    for (int r = 0; r < 8; ++r) {
      const int row = blockM + msub * 16 + lhalf * 8 + r;
      Kout[(size_t)row * kBD + ncol] = accK[t][r];
      Vout[(size_t)row * kBD + ncol] = accV[t][r];
    }
  }
}

// ============================================================
// Kernel 3: scores[b,h,n,s] = (Q[n,h,:] . K[b,s,h,:]) / 8, masked
//   grid = (S/256, B*H), block 256; one thread -> one s, all 8 slots
// ============================================================
__global__ __launch_bounds__(256) void scores_kernel(
    const float* __restrict__ Qbuf, const float* __restrict__ Kbuf,
    const int* __restrict__ mask, float* __restrict__ Sbuf) {
  __shared__ float qs[kSLOTS * kHD];   // 8x64
  const int tid = threadIdx.x;
  const int bh = blockIdx.y, b = bh >> 3, h = bh & 7;
  for (int i = tid; i < kSLOTS * kHD; i += 256) {
    int n = i >> 6, d = i & 63;
    qs[i] = Qbuf[n * kBD + h * kHD + d];
  }
  __syncthreads();
  const int s = blockIdx.x * 256 + tid;
  const float* krow = Kbuf + ((size_t)b * kS + s) * kBD + h * kHD;
  float acc[8] = {0, 0, 0, 0, 0, 0, 0, 0};
  for (int d = 0; d < kHD; d += 4) {
    f32x4 kv = *(const f32x4*)(krow + d);
    #pragma unroll
    for (int n = 0; n < 8; ++n)
      acc[n] += qs[n*64+d]*kv[0] + qs[n*64+d+1]*kv[1]
              + qs[n*64+d+2]*kv[2] + qs[n*64+d+3]*kv[3];
  }
  const int mv = mask[b * kS + s];
  float* out = Sbuf + (size_t)bh * kSLOTS * kS + s;
  #pragma unroll
  for (int n = 0; n < 8; ++n)
    out[(size_t)n * kS] = (mv == 0) ? -3.4028235e38f : acc[n] * 0.125f;
}

// ============================================================
// Kernel 4: row softmax over s (256 rows of 4096), in place
// ============================================================
__global__ __launch_bounds__(256) void softmax_kernel(float* __restrict__ Sbuf) {
  __shared__ float red[256];
  float* row = Sbuf + (size_t)blockIdx.x * kS;
  const int tid = threadIdx.x;
  float m = -3.4028235e38f;
  for (int i = tid; i < kS; i += 256) m = fmaxf(m, row[i]);
  red[tid] = m; __syncthreads();
  for (int off = 128; off > 0; off >>= 1) {
    if (tid < off) red[tid] = fmaxf(red[tid], red[tid + off]);
    __syncthreads();
  }
  const float mx = red[0];
  __syncthreads();
  float sum = 0.f;
  for (int i = tid; i < kS; i += 256) sum += __expf(row[i] - mx);
  red[tid] = sum; __syncthreads();
  for (int off = 128; off > 0; off >>= 1) {
    if (tid < off) red[tid] += red[tid + off];
    __syncthreads();
  }
  const float inv = 1.0f / red[0];
  for (int i = tid; i < kS; i += 256) row[i] = __expf(row[i] - mx) * inv;
}

// ============================================================
// Kernel 5: out[b,n,h,d] = sum_s attn[b,h,n,s] * V[b,s,h,d]
//   grid = 256 (one block per (b,h,n)), block 256 = 4 s-partitions x 64 d
// ============================================================
__global__ __launch_bounds__(256) void attnv_kernel(
    const float* __restrict__ Sbuf, const float* __restrict__ Vbuf,
    float* __restrict__ Obuf) {
  __shared__ float red[256];
  const int tid = threadIdx.x;
  const int idx = blockIdx.x;                 // (b,h,n)
  const int b = idx >> 6, h = (idx >> 3) & 7, n = idx & 7;
  const float* arow = Sbuf + (size_t)idx * kS;
  const int d = tid & 63, sb = tid >> 6;
  float acc = 0.f;
  for (int s = sb; s < kS; s += 4)
    acc += arow[s] * Vbuf[((size_t)b * kS + s) * kBD + h * kHD + d];
  red[tid] = acc;
  __syncthreads();
  if (sb == 0)
    Obuf[((size_t)b * kSLOTS + n) * kBD + h * kHD + d] =
        red[d] + red[64 + d] + red[128 + d] + red[192 + d];
}

// ============================================================
// Kernel 6: Y[b,n,o] = sum_d out[b,n,d] * Wo[o,d]  (32 x 4096)
// ============================================================
__global__ __launch_bounds__(256) void outproj_kernel(
    const float* __restrict__ Obuf, const float* __restrict__ Wo,
    float* __restrict__ Y) {
  __shared__ float orow[kBD];
  const int tid = threadIdx.x;
  const int bn = blockIdx.x;                 // 0..31
  orow[tid]       = Obuf[(size_t)bn * kBD + tid];
  orow[tid + 256] = Obuf[(size_t)bn * kBD + tid + 256];
  __syncthreads();
  for (int k = 0; k < 16; ++k) {
    const int o = k * 256 + tid;
    const float* w = Wo + (size_t)o * kBD;
    float acc = 0.f;
    for (int dd = 0; dd < kBD; dd += 4) {
      f32x4 wv = *(const f32x4*)(w + dd);
      acc += orow[dd]*wv[0] + orow[dd+1]*wv[1] + orow[dd+2]*wv[2] + orow[dd+3]*wv[3];
    }
    Y[(size_t)bn * kHID + o] = acc;
  }
}

// ============================================================
extern "C" void kernel_launch(void* const* d_in, const int* in_sizes, int n_in,
                              void* d_out, int out_size, void* d_ws, size_t ws_size,
                              hipStream_t stream) {
  const float* H    = (const float*)d_in[0];   // (4,4096,4096)
  const int*   mask = (const int*)  d_in[1];   // (4,4096)
  const float* mem  = (const float*)d_in[2];   // (8,4096)
  const float* Wq   = (const float*)d_in[3];   // (512,4096)
  const float* Wk   = (const float*)d_in[4];   // (512,4096)
  const float* Wv   = (const float*)d_in[5];   // (512,4096)
  const float* Wo   = (const float*)d_in[6];   // (4096,512)
  float* Y = (float*)d_out;                    // (4,8,4096)

  // workspace carve-up (floats)
  float* ws   = (float*)d_ws;
  float* Qbuf = ws;                                   // 8*512
  float* Kbuf = Qbuf + (size_t)kSLOTS * kBD;          // 16384*512
  float* Vbuf = Kbuf + (size_t)kM * kBD;              // 16384*512
  float* Sbuf = Vbuf + (size_t)kM * kBD;              // 4*8*8*4096
  float* Obuf = Sbuf + (size_t)kB * kHEADS * kSLOTS * kS;   // 32*512

  qproj_kernel<<<16, 256, 0, stream>>>(mem, Wq, Qbuf);

  dim3 gkv(kM / MBLK, kBD / NBLK);   // (256, 4)
  kvproj_wmma_kernel<<<gkv, 256, 0, stream>>>(H, Wk, Wv, Kbuf, Vbuf);

  dim3 gsc(kS / 256, kB * kHEADS);   // (16, 32)
  scores_kernel<<<gsc, 256, 0, stream>>>(Qbuf, Kbuf, mask, Sbuf);

  softmax_kernel<<<kB * kHEADS * kSLOTS, 256, 0, stream>>>(Sbuf);

  attnv_kernel<<<kB * kHEADS * kSLOTS, 256, 0, stream>>>(Sbuf, Vbuf, Obuf);

  outproj_kernel<<<kB * kSLOTS, 256, 0, stream>>>(Obuf, Wo, Y);
}
